// GCN_30399778521512
// MI455X (gfx1250) — matrix operand order repaired and checked by reading, not compiled, not run on hardware
//
#include <hip/hip_runtime.h>
#include <hip/hip_bf16.h>

typedef __attribute__((ext_vector_type(16))) __bf16 v16bf;
typedef __attribute__((ext_vector_type(8)))  __bf16 v8bf;
typedef __attribute__((ext_vector_type(8)))  float  v8f;

// fp32 -> bf16 (round to nearest even)
__device__ __forceinline__ __bf16 tobf(float f) {
  unsigned int u = __float_as_uint(f);
  u += 0x7FFFu + ((u >> 16) & 1u);
  unsigned short s = (unsigned short)(u >> 16);
  return __builtin_bit_cast(__bf16, s);
}

// ---------------------------------------------------------------------------
// WMMA GEMM: Hout[M,D] (fp32) = Ab[M,K] (bf16, row-major) @ Bt[D,K]^T (bf16)
// Each wave computes a 16x64 strip: one A fragment feeds 4 WMMA tiles.
// Inner loop: 2 b128 A-loads + 8 b128 B-loads + 4 v_wmma. No conversion VALU.
// ---------------------------------------------------------------------------
__global__ __launch_bounds__(128) void gemm_bf16_wmma(
    const __bf16* __restrict__ Ab, const __bf16* __restrict__ Bt,
    float* __restrict__ Hout, int K, int D, int cgShift, int tasks)
{
  const int lane = threadIdx.x & 31;
  const int wave = threadIdx.x >> 5;
  const int task = blockIdx.x * 4 + wave;
  if (task >= tasks) return;
  const int rowBlk = task >> cgShift;              // 16-row tile index
  const int cg     = task & ((1 << cgShift) - 1);  // 64-col group index
  const int m16 = lane & 15;
  const int hi  = lane >> 4;                        // lane group 0/1

  const int row   = (rowBlk << 4) + m16;
  const int nLane = (cg << 6) + m16;                // column of tile j is nLane + 16*j

  // A: lanes 0-15 need K {0..15}+k0 of row `row`; lanes 16-31: {16..31}
  const __bf16* ap = Ab + (size_t)row * K + (hi << 4);
  // B tile j: lane holds column nLane+16j, K = k0 + hi*16 .. +15 (contiguous in Bt)
  const __bf16* bp = Bt + (size_t)nLane * K + (hi << 4);
  const size_t bTileStride = (size_t)16 * K;        // between the 4 column tiles

  v8f c0 = {}, c1 = {}, c2 = {}, c3 = {};
  for (int k0 = 0; k0 < K; k0 += 32) {
    const v8bf alo = *(const v8bf*)(ap + k0);
    const v8bf ahi = *(const v8bf*)(ap + k0 + 8);
    const v16bf a = __builtin_shufflevector(alo, ahi,
        0, 1, 2, 3, 4, 5, 6, 7, 8, 9, 10, 11, 12, 13, 14, 15);

    const __bf16* b0p = bp + k0;
    const v16bf b0 = __builtin_shufflevector(
        *(const v8bf*)(b0p), *(const v8bf*)(b0p + 8),
        0, 1, 2, 3, 4, 5, 6, 7, 8, 9, 10, 11, 12, 13, 14, 15);
    c0 = __builtin_amdgcn_wmma_f32_16x16x32_bf16(false, a, false, b0, (short)0, c0, false, false);

    const __bf16* b1p = b0p + bTileStride;
    const v16bf b1 = __builtin_shufflevector(
        *(const v8bf*)(b1p), *(const v8bf*)(b1p + 8),
        0, 1, 2, 3, 4, 5, 6, 7, 8, 9, 10, 11, 12, 13, 14, 15);
    c1 = __builtin_amdgcn_wmma_f32_16x16x32_bf16(false, a, false, b1, (short)0, c1, false, false);

    const __bf16* b2p = b1p + bTileStride;
    const v16bf b2 = __builtin_shufflevector(
        *(const v8bf*)(b2p), *(const v8bf*)(b2p + 8),
        0, 1, 2, 3, 4, 5, 6, 7, 8, 9, 10, 11, 12, 13, 14, 15);
    c2 = __builtin_amdgcn_wmma_f32_16x16x32_bf16(false, a, false, b2, (short)0, c2, false, false);

    const __bf16* b3p = b2p + bTileStride;
    const v16bf b3 = __builtin_shufflevector(
        *(const v8bf*)(b3p), *(const v8bf*)(b3p + 8),
        0, 1, 2, 3, 4, 5, 6, 7, 8, 9, 10, 11, 12, 13, 14, 15);
    c3 = __builtin_amdgcn_wmma_f32_16x16x32_bf16(false, a, false, b3, (short)0, c3, false, false);
  }

  // store: c_j[r] -> (row rowBlk*16 + hi*8 + r, col cg*64 + 16*j + m16)
  float* op = Hout + (size_t)((rowBlk << 4) + (hi << 3)) * D + (cg << 6) + m16;
  #pragma unroll
  for (int r = 0; r < 8; ++r) {
    float* orow = op + (size_t)r * D;
    orow[0]  = c0[r];
    orow[16] = c1[r];
    orow[32] = c2[r];
    orow[48] = c3[r];
  }
}

// ---------------------------------------------------------------------------
// Weight transpose + bf16 convert: Wt[d*K+k] = bf16(W[k*D+d])   (tiny)
// ---------------------------------------------------------------------------
__global__ __launch_bounds__(256) void wtrans_kernel(
    const float* __restrict__ W, __bf16* __restrict__ Wt, int K, int D)
{
  int idx = blockIdx.x * 256 + threadIdx.x;
  if (idx >= K * D) return;
  int d = idx / K;
  int k = idx - d * K;
  Wt[idx] = tobf(W[(size_t)k * D + d]);
}

// fp32 -> bf16 bulk convert (for x)
__global__ __launch_bounds__(256) void cvt_bf16_kernel(
    const float* __restrict__ in, __bf16* __restrict__ out, long long n)
{
  long long i = (long long)blockIdx.x * 256 + threadIdx.x;
  if (i < n) out[i] = tobf(in[i]);
}

// ---------------------------------------------------------------------------
// Degree / norm precompute
// ---------------------------------------------------------------------------
__global__ void init_deg_kernel(float* deg, int n) {
  int i = blockIdx.x * blockDim.x + threadIdx.x;
  if (i < n) deg[i] = 1.0f;                 // self-loop
}
__global__ void deg_accum_kernel(const int* __restrict__ dst, float* deg, int e) {
  int i = blockIdx.x * blockDim.x + threadIdx.x;
  if (i < e) atomicAdd(&deg[dst[i]], 1.0f);
}
__global__ void deg_rsqrt_kernel(float* deg, int n) {
  int i = blockIdx.x * blockDim.x + threadIdx.x;
  if (i < n) deg[i] = rsqrtf(deg[i]);       // in-place: deg -> deg^{-1/2}
}
__global__ void enorm_kernel(const int* __restrict__ src, const int* __restrict__ dst,
                             const float* __restrict__ dis, float* __restrict__ en, int e) {
  int i = blockIdx.x * blockDim.x + threadIdx.x;
  if (i < e) en[i] = dis[src[i]] * dis[dst[i]];
}

// ---------------------------------------------------------------------------
// Edge scatter: agg[dst, :] += h[src, :] * enorm[e]   (float4 gather, f32 atomics)
// chunk fastest-varying -> coalesced; h (<=102 MB) is L2-resident (192 MB L2).
// ---------------------------------------------------------------------------
__global__ __launch_bounds__(256) void scatter_kernel(
    const float* __restrict__ h, const int* __restrict__ src,
    const int* __restrict__ dst, const float* __restrict__ enorm,
    float* __restrict__ agg, int D, int chunkShift, long long total)
{
  long long idx = (long long)blockIdx.x * blockDim.x + threadIdx.x;
  if (idx >= total) return;
  const int c = (int)(idx & ((1 << chunkShift) - 1));
  const int e = (int)(idx >> chunkShift);
  const int s = src[e];
  const int d = dst[e];
  const float w = enorm[e];
  const float4 v = *(const float4*)(h + (size_t)s * D + (c << 2));
  float* o = agg + (size_t)d * D + (c << 2);
  atomicAdd(o + 0, v.x * w);
  atomicAdd(o + 1, v.y * w);
  atomicAdd(o + 2, v.z * w);
  atomicAdd(o + 3, v.w * w);
}

// ---------------------------------------------------------------------------
// Fused epilogue: v = relu(agg + h*dis^2 + bias); write back; per-column
// sum/sumsq in registers, one atomic per block per column. blockDim.x == D.
// ---------------------------------------------------------------------------
__global__ void epilogue_kernel(const float* __restrict__ h,
    float* __restrict__ agg, const float* __restrict__ bias,
    const float* __restrict__ dis, float* __restrict__ colsum,
    float* __restrict__ colsq, int Nn, int D, int rowsPerBlock)
{
  const int f = threadIdx.x;
  const int r0 = blockIdx.x * rowsPerBlock;
  int r1 = r0 + rowsPerBlock; if (r1 > Nn) r1 = Nn;
  const float b = bias[f];
  float s = 0.f, sq = 0.f;
  for (int r = r0; r < r1; ++r) {
    const float di = dis[r];
    const size_t off = (size_t)r * D + f;
    float v = agg[off] + h[off] * (di * di) + b;
    v = fmaxf(v, 0.f);
    agg[off] = v;
    s += v; sq += v * v;
  }
  atomicAdd(&colsum[f], s);
  atomicAdd(&colsq[f], sq);
}

// mean/var -> affine scale/shift (one tiny block)
__global__ void finalize_kernel(const float* __restrict__ colsum,
    const float* __restrict__ colsq, const float* __restrict__ g,
    const float* __restrict__ beta, float* __restrict__ scale,
    float* __restrict__ shift, int Nn, int D)
{
  int f = threadIdx.x;
  if (f >= D) return;
  const float invN = 1.0f / (float)Nn;
  const float mu  = colsum[f] * invN;
  const float var = colsq[f] * invN - mu * mu;
  const float sc  = g[f] * rsqrtf(var + 1e-5f);
  scale[f] = sc;
  shift[f] = beta[f] - mu * sc;
}

// normalize: fp32 in place (for classifier) + bf16 copy (for next WMMA GEMM)
__global__ __launch_bounds__(256) void normalize_kernel(
    float* __restrict__ buf, __bf16* __restrict__ outb,
    const float* __restrict__ scale, const float* __restrict__ shift,
    long long total, int Dmask)
{
  long long i = (long long)blockIdx.x * blockDim.x + threadIdx.x;
  if (i >= total) return;
  const int f = (int)(i & Dmask);
  const float v = buf[i] * scale[f] + shift[f];
  buf[i] = v;
  outb[i] = tobf(v);
}

// ---------------------------------------------------------------------------
// Classifier: out[N,10] = h[N,64] @ Wc[64,10] + bc  (LDS-staged weights)
// ---------------------------------------------------------------------------
__global__ __launch_bounds__(256) void classifier_kernel(
    const float* __restrict__ hin, const float* __restrict__ Wc,
    const float* __restrict__ bc, float* __restrict__ out, int Nn)
{
  __shared__ float sW[64 * 10];
  __shared__ float sB[10];
  for (int i = threadIdx.x; i < 640; i += 256) sW[i] = Wc[i];
  if (threadIdx.x < 10) sB[threadIdx.x] = bc[threadIdx.x];
  __syncthreads();
  const int row = blockIdx.x * 256 + threadIdx.x;
  if (row >= Nn) return;
  float acc[10];
  #pragma unroll
  for (int j = 0; j < 10; ++j) acc[j] = sB[j];
  const float* xp = hin + (size_t)row * 64;
  for (int k = 0; k < 64; ++k) {
    const float xv = xp[k];
    #pragma unroll
    for (int j = 0; j < 10; ++j) acc[j] += xv * sW[k * 10 + j];
  }
  float* op = out + (size_t)row * 10;
  #pragma unroll
  for (int j = 0; j < 10; ++j) op[j] = acc[j];
}

// ---------------------------------------------------------------------------
// Host orchestration
// ---------------------------------------------------------------------------
static inline unsigned gblk(long long total, int bs) {
  return (unsigned)((total + bs - 1) / bs);
}

extern "C" void kernel_launch(void* const* d_in, const int* in_sizes, int n_in,
                              void* d_out, int out_size, void* d_ws, size_t ws_size,
                              hipStream_t stream) {
  const float* x   = (const float*)d_in[0];
  const int*   ei  = (const int*)d_in[1];
  const float* W[4]  = {(const float*)d_in[2],  (const float*)d_in[6],
                        (const float*)d_in[10], (const float*)d_in[14]};
  const float* Bv[4] = {(const float*)d_in[3],  (const float*)d_in[7],
                        (const float*)d_in[11], (const float*)d_in[15]};
  const float* Gv[4] = {(const float*)d_in[4],  (const float*)d_in[8],
                        (const float*)d_in[12], (const float*)d_in[16]};
  const float* Bt4[4]= {(const float*)d_in[5],  (const float*)d_in[9],
                        (const float*)d_in[13], (const float*)d_in[17]};
  const float* Wc = (const float*)d_in[18];
  const float* bc = (const float*)d_in[19];

  const int Nn = in_sizes[0] / 512;     // 100000
  const int Ee = in_sizes[1] / 2;       // 1600000
  const int* src = ei;
  const int* dst = ei + Ee;

  // workspace layout (floats), 256-element aligned regions
  float* ws = (float*)d_ws;
  size_t off = 0;
  float* dis    = ws + off; off += ((size_t)Nn + 255) & ~(size_t)255;
  float* enorm  = ws + off; off += ((size_t)Ee + 255) & ~(size_t)255;
  float* colsum = ws + off; off += 256;
  float* colsq  = ws + off; off += 256;
  float* scale  = ws + off; off += 256;
  float* shift  = ws + off; off += 256;
  float* bufH   = ws + off; off += (size_t)Nn * 256;   // GEMM output (fp32)
  float* bufA   = ws + off; off += (size_t)Nn * 256;   // agg / relu'd (fp32)
  __bf16* Xb    = (__bf16*)(ws + off); off += (size_t)Nn * 256;  // N*512 bf16 acts
  __bf16* Wt    = (__bf16*)(ws + off); off += 32768;   // 65536 bf16 weights

  // ---- degree + per-edge norm (layer-invariant) ----
  init_deg_kernel<<<gblk(Nn, 256), 256, 0, stream>>>(dis, Nn);
  deg_accum_kernel<<<gblk(Ee, 256), 256, 0, stream>>>(dst, dis, Ee);
  deg_rsqrt_kernel<<<gblk(Nn, 256), 256, 0, stream>>>(dis, Nn);
  enorm_kernel<<<gblk(Ee, 256), 256, 0, stream>>>(src, dst, dis, enorm, Ee);

  // ---- convert x to bf16 once ----
  cvt_bf16_kernel<<<gblk((long long)Nn * 512, 256), 256, 0, stream>>>(
      x, Xb, (long long)Nn * 512);

  // ---- 4 GCN layers ----
  const int Kdim[4] = {512, 128, 256, 128};
  const int Ddim[4] = {128, 256, 128, 64};
  for (int l = 0; l < 4; ++l) {
    const int K = Kdim[l], D = Ddim[l];
    // transpose+convert weights (tiny)
    wtrans_kernel<<<gblk(K * D, 256), 256, 0, stream>>>(W[l], Wt, K, D);
    // WMMA GEMM, 16x64 strip per wave
    const int cgShift = (D == 256) ? 2 : (D == 128) ? 1 : 0;
    const int tasks = (Nn / 16) << cgShift;
    gemm_bf16_wmma<<<gblk(tasks, 4), 128, 0, stream>>>(
        Xb, Wt, bufH, K, D, cgShift, tasks);
    // zero agg + stats
    hipMemsetAsync(bufA, 0, (size_t)Nn * D * sizeof(float), stream);
    hipMemsetAsync(colsum, 0, 512 * sizeof(float), stream);
    // edge scatter
    const int chunkShift = (D == 256) ? 6 : (D == 128) ? 5 : 4;
    const long long total = (long long)Ee << chunkShift;
    scatter_kernel<<<gblk(total, 256), 256, 0, stream>>>(
        bufH, src, dst, enorm, bufA, D, chunkShift, total);
    // bias + self-loop + relu + column stats
    const int rowsPerBlock = 128;
    epilogue_kernel<<<gblk(Nn, rowsPerBlock), D, 0, stream>>>(
        bufH, bufA, Bv[l], dis, colsum, colsq, Nn, D, rowsPerBlock);
    finalize_kernel<<<1, 256, 0, stream>>>(colsum, colsq, Gv[l], Bt4[l],
                                           scale, shift, Nn, D);
    const long long tot2 = (long long)Nn * D;
    normalize_kernel<<<gblk(tot2, 256), 256, 0, stream>>>(
        bufA, Xb, scale, shift, tot2, D - 1);
  }

  // ---- classifier (reads fp32 normalized h4) ----
  classifier_kernel<<<gblk(Nn, 256), 256, 0, stream>>>(
      bufA, Wc, bc, (float*)d_out, Nn);
}
